// NetBA_9466107920964
// MI455X (gfx1250) — compile-verified, matrix operand order checked by reading, not run on hardware
//
#include <hip/hip_runtime.h>
#include <hip/hip_bf16.h>
#include <math.h>

typedef __attribute__((ext_vector_type(16))) __bf16 v16bf;
typedef __attribute__((ext_vector_type(8)))  float  v8f;
typedef __attribute__((ext_vector_type(8)))  unsigned short us8;

#define GRAPHS 64
#define BN_EPS 1e-5f

static __device__ __forceinline__ unsigned short f2bf(float f) {
  union { float f; unsigned u; } v; v.f = f;
  unsigned r = v.u + 0x7FFFu + ((v.u >> 16) & 1u);
  return (unsigned short)(r >> 16);
}

__global__ void zero_f32(float* p, int n) {
  int i = blockIdx.x * blockDim.x + threadIdx.x;
  if (i < n) p[i] = 0.0f;
}

__global__ void copy_f32v4(float4* __restrict__ dst, const float4* __restrict__ src, int n4) {
  int i = blockIdx.x * blockDim.x + threadIdx.x;
  if (i < n4) dst[i] = src[i];
}

// agg[dst] += feat[src] per edge; one thread handles 4 features (float4 gather).
// total = E * (F/4), F = 1<<fshift.
__global__ void scatter_add4(float* __restrict__ agg, const float* __restrict__ feat,
                             const int* __restrict__ esrc, const int* __restrict__ edst,
                             int total, int fshift) {
  int i = blockIdx.x * blockDim.x + threadIdx.x;
  if (i >= total) return;
  int fq = 1 << (fshift - 2);
  int e  = i >> (fshift - 2);
  int f4 = (i & (fq - 1)) << 2;
  int s = esrc[e];
  int d = edst[e];
  const float4 v = *(const float4*)&feat[(s << fshift) + f4];
  float* a = &agg[(d << fshift) + f4];
  atomicAdd(a + 0, v.x);
  atomicAdd(a + 1, v.y);
  atomicAdd(a + 2, v.z);
  atomicAdd(a + 3, v.w);
}

// Pack fp32 W[K x NOUT] (row-major) into bf16 WMMA-B-fragment order:
// per (k-step kt, col-tile nt): 32 lanes x 16 contiguous bf16 each.
__global__ void pack_w_bf16(unsigned short* __restrict__ wp, const float* __restrict__ W,
                            int K, int NOUT) {
  int idx = blockIdx.x * blockDim.x + threadIdx.x;
  if (idx >= K * NOUT) return;
  int k = idx / NOUT, n = idx % NOUT;
  int kt = k >> 5, kr = k & 31;
  int nt = n >> 4, nr = n & 15;
  int lane = ((kr >> 4) << 4) | nr;
  int i = kr & 15;
  wp[(((kt * (NOUT >> 4) + nt) * 32 + lane) << 4) + i] = f2bf(W[idx]);
}

// C[M x NOUT] = A[M x K](fp32 -> bf16 via LDS) @ Wp(bf16 packed) + bias.
// Block: 256 threads = 8 waves; block tile = 64 rows x (128*NT) cols.
// Each wave: 4 stacked M-tiles x NT N-tiles (4*NT accumulators); the A fragment
// is loaded from LDS once per M-tile and reused across NT B fragments.
// STATS: fused BN column sum / sum-of-squares via LDS reduction + global atomics.
template<int K, int NOUT, bool RELU, bool STATS, int NT>
__global__ void __launch_bounds__(256) gemm_bf16(float* __restrict__ C,
                                                 const float* __restrict__ A,
                                                 const unsigned short* __restrict__ Wp,
                                                 const float* __restrict__ bias,
                                                 float* __restrict__ gsum,
                                                 float* __restrict__ gsq,
                                                 int nrows) {
  __shared__ __align__(32) unsigned short a_lds[4][16 * 32];
  __shared__ float s_sum[256], s_sq[256];
  const int t      = threadIdx.x;
  const int lane   = t & 31;
  const int wave   = t >> 5;
  const int baseM  = blockIdx.x * 64;
  const int tileN0 = (blockIdx.y * 8 + wave) * NT;

  float bv[NT];
  int   col[NT];
#pragma unroll
  for (int nt = 0; nt < NT; ++nt) {
    col[nt] = (tileN0 + nt) * 16 + (lane & 15);
    bv[nt]  = bias[col[nt]];
  }
  if (STATS && t < 128 * NT) { s_sum[t] = 0.0f; s_sq[t] = 0.0f; }

  // Staging decomposition: thread loads 8 consecutive K-elements of one row.
  const int rowIdx = t >> 2;          // 0..63
  const int rowIn  = rowIdx & 15;
  const int mtS    = rowIdx >> 4;     // which row-tile this thread stages into
  const int kkBase = (t & 3) << 3;    // 0,8,16,24
  const int r      = baseM + rowIdx;
  const int ldsOff = (((kkBase >> 4) << 4) + rowIn) * 16 + (kkBase & 15);

  v8f acc[4][NT] = {};
#pragma unroll
  for (int kt = 0; kt < K / 32; ++kt) {
    __syncthreads();
    float4 v0 = make_float4(0.f, 0.f, 0.f, 0.f), v1 = v0;
    if (r < nrows) {
      const float* ap = &A[r * K + kt * 32 + kkBase];
      v0 = *(const float4*)ap;
      v1 = *(const float4*)(ap + 4);
    }
    us8 o;
    o[0] = f2bf(v0.x); o[1] = f2bf(v0.y); o[2] = f2bf(v0.z); o[3] = f2bf(v0.w);
    o[4] = f2bf(v1.x); o[5] = f2bf(v1.y); o[6] = f2bf(v1.z); o[7] = f2bf(v1.w);
    *(us8*)&a_lds[mtS][ldsOff] = o;
    __syncthreads();
    v16bf bf[NT];
#pragma unroll
    for (int nt = 0; nt < NT; ++nt)
      bf[nt] = *(const v16bf*)&Wp[((kt * (NOUT >> 4) + tileN0 + nt) * 32 + lane) * 16];
#pragma unroll
    for (int mt = 0; mt < 4; ++mt) {
      v16bf af = *(const v16bf*)&a_lds[mt][lane * 16];
#pragma unroll
      for (int nt = 0; nt < NT; ++nt)
        acc[mt][nt] = __builtin_amdgcn_wmma_f32_16x16x32_bf16(
            false, af, false, bf[nt], (short)0, acc[mt][nt], false, false);
    }
  }

#pragma unroll
  for (int mt = 0; mt < 4; ++mt) {
    int rb = baseM + mt * 16;
    if (rb >= nrows) break;          // block-uniform (nrows % 16 == 0)
#pragma unroll
    for (int nt = 0; nt < NT; ++nt) {
      float s = 0.0f, q = 0.0f;
#pragma unroll
      for (int rr = 0; rr < 8; ++rr) {
        int row = rb + ((lane < 16) ? rr : (rr + 8));
        float v = acc[mt][nt][rr] + bv[nt];
        if (STATS) { s += v; q += v * v; }
        if (RELU) v = fmaxf(v, 0.0f);
        C[row * NOUT + col[nt]] = v;
      }
      if (STATS) {
        int cb = (wave * NT + nt) * 16 + (lane & 15);
        atomicAdd(&s_sum[cb], s);
        atomicAdd(&s_sq [cb], q);
      }
    }
  }
  if (STATS) {
    __syncthreads();
    if (t < 128 * NT) {
      atomicAdd(&gsum[blockIdx.y * 128 * NT + t], s_sum[t]);
      atomicAdd(&gsq [blockIdx.y * 128 * NT + t], s_sq[t]);
    }
  }
}

__global__ void bn_finalize(float* __restrict__ scale, float* __restrict__ shift,
                            const float* __restrict__ sums, const float* __restrict__ sqs,
                            const float* __restrict__ gamma, const float* __restrict__ beta,
                            int NOUT, float invN) {
  int c = blockIdx.x * blockDim.x + threadIdx.x;
  if (c >= NOUT) return;
  float mean = sums[c] * invN;
  float var  = sqs[c] * invN - mean * mean;
  float sc   = gamma[c] * rsqrtf(var + BN_EPS);
  scale[c] = sc;
  shift[c] = beta[c] - mean * sc;
}

// BN apply (+optional ReLU), float4, NOUT fixed at 256.
// Optionally duplicates the result into `dup` (seed for next layer's scatter-add).
__global__ void bn_apply256(float4* __restrict__ h, float4* __restrict__ dup,
                            const float* __restrict__ scale, const float* __restrict__ shift,
                            int total4, int relu) {
  int i = blockIdx.x * blockDim.x + threadIdx.x;
  if (i >= total4) return;
  int cb = (i << 2) & 255;
  float4 sc = *(const float4*)&scale[cb];
  float4 sh = *(const float4*)&shift[cb];
  float4 v = h[i];
  v.x = v.x * sc.x + sh.x;
  v.y = v.y * sc.y + sh.y;
  v.z = v.z * sc.z + sh.z;
  v.w = v.w * sc.w + sh.w;
  if (relu) {
    v.x = fmaxf(v.x, 0.f); v.y = fmaxf(v.y, 0.f);
    v.z = fmaxf(v.z, 0.f); v.w = fmaxf(v.w, 0.f);
  }
  h[i] = v;
  if (dup) dup[i] = v;
}

// Final linear 128->1 + sigmoid, fused with graph mean-pool accumulation.
__global__ void mlp2_pool(const float* __restrict__ h, const float* __restrict__ W,
                          const float* __restrict__ b, const int* __restrict__ batch,
                          float* __restrict__ pool, int rows) {
  int n = blockIdx.x * blockDim.x + threadIdx.x;
  if (n >= rows) return;
  const float4* h4 = (const float4*)&h[n * 128];
  const float4* w4 = (const float4*)W;
  float acc = b[0];
#pragma unroll
  for (int k = 0; k < 32; ++k) {
    float4 a = h4[k], w = w4[k];
    acc += a.x * w.x + a.y * w.y + a.z * w.z + a.w * w.w;
  }
  float sg = 1.0f / (1.0f + __expf(-acc));
  int g = batch[n];
  atomicAdd(&pool[g], sg);
  atomicAdd(&pool[GRAPHS + g], 1.0f);
}

__global__ void pool_finalize(float* __restrict__ out, const float* __restrict__ pool) {
  int g = threadIdx.x;
  if (g < GRAPHS) out[g] = pool[g] / fmaxf(pool[GRAPHS + g], 1.0f);
}

extern "C" void kernel_launch(void* const* d_in, const int* in_sizes, int n_in,
                              void* d_out, int out_size, void* d_ws, size_t ws_size,
                              hipStream_t stream) {
  const float* x    = (const float*)d_in[0];
  const int*   ei   = (const int*)d_in[1];
  const int*   batch= (const int*)d_in[2];
  const float* W1   = (const float*)d_in[3];  const float* b1  = (const float*)d_in[4];
  const float* g1   = (const float*)d_in[5];  const float* be1 = (const float*)d_in[6];
  const float* W2   = (const float*)d_in[7];  const float* b2  = (const float*)d_in[8];
  const float* g2   = (const float*)d_in[9];  const float* be2 = (const float*)d_in[10];
  const float* W3   = (const float*)d_in[11]; const float* b3  = (const float*)d_in[12];
  const float* g3   = (const float*)d_in[13]; const float* be3 = (const float*)d_in[14];
  const float* Wl1  = (const float*)d_in[15]; const float* bl1 = (const float*)d_in[16];
  const float* Wl2  = (const float*)d_in[17]; const float* bl2 = (const float*)d_in[18];

  const int nrows = in_sizes[0] / 64;   // 50000
  const int E     = in_sizes[1] / 2;    // 800000

  // Workspace layout (floats)
  const size_t NB = (size_t)nrows * 256;
  float* f     = (float*)d_ws;
  float* bufA  = f;
  float* bufS  = f + NB;
  float* bufO  = f + 2 * NB;
  float* stats = f + 3 * NB;            // sums[256], sqs[256], scale[256], shift[256]
  float* pool  = stats + 1024;          // sums[64], counts[64]
  unsigned short* wp = (unsigned short*)(pool + 128);  // 256*256 bf16

  const size_t needed = (3 * NB + 1152) * sizeof(float) + 256 * 256 * sizeof(unsigned short);
  if (ws_size < needed) return;

  const int gx = (nrows + 63) / 64;     // 64-row GEMM tiles
  const float invN = 1.0f / (float)nrows;

  // ---------------- Layer 1: F_IN=64 -> DIM=256 ----------------
  {
    int n4 = nrows * 16;  // nrows*64 floats
    copy_f32v4<<<(n4 + 255) / 256, 256, 0, stream>>>((float4*)bufS, (const float4*)x, n4);
    int tot = E * 16;
    scatter_add4<<<(tot + 255) / 256, 256, 0, stream>>>(bufS, x, ei, ei + E, tot, 6);
    pack_w_bf16<<<(64 * 256 + 255) / 256, 256, 0, stream>>>(wp, W1, 64, 256);
    zero_f32<<<2, 256, 0, stream>>>(stats, 512);
    gemm_bf16<64, 256, false, true, 2><<<dim3(gx, 1), 256, 0, stream>>>(
        bufO, bufS, wp, b1, stats, stats + 256, nrows);
    bn_finalize<<<1, 256, 0, stream>>>(stats + 512, stats + 768, stats, stats + 256, g1, be1, 256, invN);
    bn_apply256<<<(nrows * 64 + 255) / 256, 256, 0, stream>>>(
        (float4*)bufO, (float4*)bufS, stats + 512, stats + 768, nrows * 64, 1);
  }
  // ---------------- Layer 2: 256 -> 256 ----------------
  {
    int tot = E * 64;
    scatter_add4<<<(tot + 255) / 256, 256, 0, stream>>>(bufS, bufO, ei, ei + E, tot, 8);
    pack_w_bf16<<<(256 * 256 + 255) / 256, 256, 0, stream>>>(wp, W2, 256, 256);
    zero_f32<<<2, 256, 0, stream>>>(stats, 512);
    gemm_bf16<256, 256, false, true, 2><<<dim3(gx, 1), 256, 0, stream>>>(
        bufA, bufS, wp, b2, stats, stats + 256, nrows);
    bn_finalize<<<1, 256, 0, stream>>>(stats + 512, stats + 768, stats, stats + 256, g2, be2, 256, invN);
    bn_apply256<<<(nrows * 64 + 255) / 256, 256, 0, stream>>>(
        (float4*)bufA, (float4*)bufS, stats + 512, stats + 768, nrows * 64, 1);
  }
  // ---------------- Layer 3: 256 -> 256 (no ReLU) ----------------
  {
    int tot = E * 64;
    scatter_add4<<<(tot + 255) / 256, 256, 0, stream>>>(bufS, bufA, ei, ei + E, tot, 8);
    pack_w_bf16<<<(256 * 256 + 255) / 256, 256, 0, stream>>>(wp, W3, 256, 256);
    zero_f32<<<2, 256, 0, stream>>>(stats, 512);
    gemm_bf16<256, 256, false, true, 2><<<dim3(gx, 1), 256, 0, stream>>>(
        bufO, bufS, wp, b3, stats, stats + 256, nrows);
    bn_finalize<<<1, 256, 0, stream>>>(stats + 512, stats + 768, stats, stats + 256, g3, be3, 256, invN);
    bn_apply256<<<(nrows * 64 + 255) / 256, 256, 0, stream>>>(
        (float4*)bufO, (float4*)nullptr, stats + 512, stats + 768, nrows * 64, 0);
  }
  // ---------------- MLP head: 256 -> 128 (ReLU) -> 1 -> sigmoid -> mean pool ----------------
  {
    pack_w_bf16<<<(256 * 128 + 255) / 256, 256, 0, stream>>>(wp, Wl1, 256, 128);
    gemm_bf16<256, 128, true, false, 1><<<dim3(gx, 1), 256, 0, stream>>>(
        bufA, bufO, wp, bl1, nullptr, nullptr, nrows);
    zero_f32<<<1, 128, 0, stream>>>(pool, 128);
    mlp2_pool<<<(nrows + 255) / 256, 256, 0, stream>>>(bufA, Wl2, bl2, batch, pool, nrows);
    pool_finalize<<<1, 64, 0, stream>>>((float*)d_out, pool);
  }
}
